// Model_55963423867283
// MI455X (gfx1250) — compile-verified, matrix operand order checked by reading, not run on hardware
//
#include <hip/hip_runtime.h>
#include <hip/hip_bf16.h>

// ---------------------------------------------------------------------------
// Problem constants (from reference)
// ---------------------------------------------------------------------------
#define BATCH 32
#define JLEN  48
#define ILEN  48
#define CTX   5
#define EDIM  128
#define HDIM  256
#define VS    32000
#define SDIM  96                 // 2*J HMM states
#define NROWS 3072               // 2 * B * J  (word rows then null rows)

typedef _Float16 v16h __attribute__((ext_vector_type(16)));
typedef float    v8f  __attribute__((ext_vector_type(8)));
typedef unsigned int u32x4 __attribute__((ext_vector_type(4)));
typedef int          i32x4 __attribute__((ext_vector_type(4)));
typedef int          i32x8 __attribute__((ext_vector_type(8)));

// ---------------------------------------------------------------------------
// Kernel A: W_vocab [K=256, N=32000] f32  ->  WvT [N, K] f16 (contiguous K per
// vocab row, i.e. exactly the per-lane B layout WMMA wants)
// ---------------------------------------------------------------------------
__global__ __launch_bounds__(256) void convert_wv(const float* __restrict__ Wv,
                                                  _Float16* __restrict__ WvT) {
  int n = blockIdx.x * 256 + threadIdx.x;      // 125*256 == 32000 exactly
  for (int kc = 0; kc < HDIM; kc += 16) {
    union { _Float16 h[16]; uint4 u[2]; } buf;
#pragma unroll
    for (int k = 0; k < 16; ++k)
      buf.h[k] = (_Float16)Wv[(size_t)(kc + k) * VS + n];   // coalesced over n
    uint4* dst = (uint4*)(WvT + (size_t)n * HDIM + kc);
    dst[0] = buf.u[0];
    dst[1] = buf.u[1];
  }
}

// ---------------------------------------------------------------------------
// Kernel B: emission MLP up to hidden layer.
// ---------------------------------------------------------------------------
__global__ __launch_bounds__(128) void hidden_kernel(
    const float* __restrict__ emb, const float* __restrict__ Wb,
    const float* __restrict__ Wh, const float* __restrict__ bh,
    const int* __restrict__ ctxW, const int* __restrict__ ctxN,
    _Float16* __restrict__ H16) {
  __shared__ float xs[CTX * EDIM];   // 640
  __shared__ float ys[EDIM];         // 128
  const int r = blockIdx.x;
  const int variant = r / (BATCH * JLEN);
  const int rr = r % (BATCH * JLEN);
  const int b = rr / JLEN, j = rr % JLEN;
  const int* __restrict__ ctx = variant ? ctxN : ctxW;
  const int t = threadIdx.x;

#pragma unroll
  for (int c = 0; c < CTX; ++c) {
    int id = ctx[b * (JLEN * CTX) + j * CTX + c];
    xs[c * EDIM + t] = emb[(size_t)id * EDIM + t];
  }
  __syncthreads();

  float y = 0.f;
  for (int c = 0; c < CTX * EDIM; ++c) y += xs[c] * Wb[c * EDIM + t];
  ys[t] = y;
  __syncthreads();

  for (int hh = t; hh < HDIM; hh += 128) {
    float acc = bh[hh];
    for (int e = 0; e < EDIM; ++e) acc += ys[e] * Wh[e * HDIM + hh];
    H16[(size_t)r * HDIM + hh] = (_Float16)tanhf(acc);
  }
}

// ---------------------------------------------------------------------------
// TDM: load one contiguous 8KB B tile (16 vocab rows x 256 K, f16) into LDS.
// 1-D descriptor: data_size=8B, tensor_dim0=tile_dim0=1024 elements.
// ---------------------------------------------------------------------------
#if __has_builtin(__builtin_amdgcn_tensor_load_to_lds)
#define HAVE_TDM 1
__device__ __forceinline__ void tdm_load_tile(const _Float16* gsrc,
                                              unsigned ldsOff) {
  unsigned long long ga = (unsigned long long)(uintptr_t)gsrc;
  u32x4 g0;
  g0.x = 1u;                                             // count=1 descriptor
  g0.y = ldsOff;                                         // lds_addr
  g0.z = (unsigned)ga;                                   // global_addr[31:0]
  g0.w = (unsigned)((ga >> 32) & 0x01ffffffu) | (2u << 30);  // addr hi + type=2
  i32x8 g1 = {0, 0, 0, 0, 0, 0, 0, 0};
  g1[0] = (3 << 16);                 // data_size = 8 bytes
  g1[1] = (int)(1024u << 16);        // tensor_dim0 = 1024
  g1[2] = (1 << 16);                 // tensor_dim1 = 1
  g1[3] = (int)(1024u << 16);        // tile_dim0 = 1024
  g1[4] = 1;                         // tile_dim1 = 1
  g1[5] = 1024;                      // tensor_dim0_stride
  i32x4 z = {0, 0, 0, 0};
#if __clang_major__ >= 23
  i32x8 z8 = {0, 0, 0, 0, 0, 0, 0, 0};
  __builtin_amdgcn_tensor_load_to_lds(g0, g1, z, z, z8, 0);
#else
  __builtin_amdgcn_tensor_load_to_lds(g0, g1, z, z, 0);
#endif
}
#else
#define HAVE_TDM 0
#endif

// ---------------------------------------------------------------------------
// Kernel C: M=3072 x K=256 x N=32000 GEMM fused with streaming row-softmax
// statistics.  48 blocks x 128 threads (4 wave32); block owns 64 rows.
//  - A tile (64 rows) staged once in LDS; A and B fragments streamed from
//    LDS inside the loop (ds_load_b128 pipelines under the WMMAs).
//  - B tile (8KB contiguous) double-buffered in LDS via the Tensor Data
//    Mover (wave 0 issues, s_wait_tensorcnt).  CRITICAL: the TDM intrinsic
//    is not modeled as a store to lB, so lB's address is escaped into an
//    asm that clobbers memory each iteration — otherwise LLVM folds the
//    never-visibly-stored LDS loads to undef (the v_nop/serialization bug
//    of rounds 2-4).
//  - Single accumulator chain: D->C accumulation is hazard-free (ISA 7.12.1
//    penalizes only D->A/B RAW).
//  - Per-lane online (max, sum-exp); one 16-lane log-sum-exp combine at end.
// ---------------------------------------------------------------------------
__global__ __launch_bounds__(128) void vocab_gemm_stats(
    const _Float16* __restrict__ H16, const _Float16* __restrict__ WvT,
    const float* __restrict__ bv, float* __restrict__ stats) {
  __shared__ __align__(16) _Float16 lA[64 * HDIM];      // 32 KB
  __shared__ __align__(16) _Float16 lB[2][16 * HDIM];   // 16 KB double buffer
  const int tid  = threadIdx.x;
  const int wave = tid >> 5;
  const int lane = tid & 31;
  const int mn   = lane & 15;   // A row within subtile == B column
  const int kh   = lane >> 4;   // K-half selector within a 32-wide chunk
  const int rowBase = blockIdx.x * 64;

  // Generic pointers to the LDS buffers; escaped into the asm fences below
  // so the compiler must treat them as written by the TDM.
  _Float16* lAp = &lA[0];
  _Float16* lBp = &lB[0][0];
  // TDM descriptor lds_addr: low 32 bits of the generic LDS address are the
  // workgroup-relative LDS byte offset (flat aperture mapping).
  const unsigned LDS_B0 = (unsigned)(uintptr_t)&lB[0][0];
  const unsigned LDS_B1 = (unsigned)(uintptr_t)&lB[1][0];

  // ---- stage A tile in LDS: 64 rows x 256 f16 = 2048 uint4 ----
  {
    const uint4* src = (const uint4*)(H16 + (size_t)rowBase * HDIM);
    uint4* dst = (uint4*)lA;
#pragma unroll
    for (int it = 0; it < 16; ++it) dst[it * 128 + tid] = src[it * 128 + tid];
  }

  float lmax[8], lsum[8];
#pragma unroll
  for (int v = 0; v < 8; ++v) { lmax[v] = -3.0e38f; lsum[v] = 0.f; }

  const int NT = VS / 16;     // 2000

  // prologue: tile 0 -> buffer 0
#if HAVE_TDM
  if (wave == 0) {
    tdm_load_tile(WvT, LDS_B0);
    __builtin_amdgcn_s_wait_tensorcnt(0);
  }
#else
  {
    const uint4* bsrc = (const uint4*)WvT;
    uint4* bdst = (uint4*)&lB[0][0];
#pragma unroll
    for (int it = 0; it < 4; ++it) bdst[it * 128 + tid] = bsrc[it * 128 + tid];
  }
#endif
  __syncthreads();
  // Escape lA/lB and clobber memory: TDM wrote LDS behind the compiler's back.
  asm volatile("" : : "s"(lAp), "s"(lBp) : "memory");

  for (int nt = 0; nt < NT; ++nt) {
    const int cur = nt & 1;
    // kick off next tile into the other buffer (overlaps with compute)
    if (nt + 1 < NT) {
#if HAVE_TDM
      if (wave == 0)
        tdm_load_tile(WvT + (size_t)(nt + 1) * 16 * HDIM,
                      cur ? LDS_B0 : LDS_B1);
#else
      const uint4* bsrc = (const uint4*)(WvT + (size_t)(nt + 1) * 16 * HDIM);
      uint4* bdst = (uint4*)&lB[cur ^ 1][0];
#pragma unroll
      for (int it = 0; it < 4; ++it) bdst[it * 128 + tid] = bsrc[it * 128 + tid];
#endif
    }

    // ---- 16x16 tile GEMM over K=256: 8 chained WMMAs, operands streamed
    //      from LDS (loads pipeline under WMMAs) ----
    v8f acc = {};
    const _Float16* Arow = lA + (wave * 16 + mn) * HDIM;
    const _Float16* Brow = &lB[cur][0] + mn * HDIM;
#pragma unroll
    for (int kk = 0; kk < 8; ++kk) {
      v16h a, b;
      const _Float16* ap = Arow + kk * 32 + kh * 8;
      const _Float16* bp = Brow + kk * 32 + kh * 8;
#pragma unroll
      for (int e = 0; e < 8; ++e) {
        a[e] = ap[e];  a[8 + e] = ap[16 + e];
        b[e] = bp[e];  b[8 + e] = bp[16 + e];
      }
      acc = __builtin_amdgcn_wmma_f32_16x16x32_f16(
          false, a, false, b, (short)0, acc, false, false);
    }

    // per-lane online softmax stats (no cross-lane traffic here)
    const float bias = bv[nt * 16 + mn];
#pragma unroll
    for (int v = 0; v < 8; ++v) {
      float x = acc[v] + bias;
      float nm = fmaxf(lmax[v], x);
      lsum[v] = lsum[v] * __expf(lmax[v] - nm) + __expf(x - nm);
      lmax[v] = nm;
    }

#if HAVE_TDM
    if (wave == 0) __builtin_amdgcn_s_wait_tensorcnt(0);
#endif
    __syncthreads();
    // New TDM data landed in the other buffer; forbid folding/hoisting.
    asm volatile("" : : "s"(lAp), "s"(lBp) : "memory");
  }

  // final cross-lane log-sum-exp combine within each 16-lane half
#pragma unroll
  for (int v = 0; v < 8; ++v) {
    float m = lmax[v], s = lsum[v];
#pragma unroll
    for (int mask = 1; mask <= 8; mask <<= 1) {
      float om = __shfl_xor(m, mask, 32);
      float os = __shfl_xor(s, mask, 32);
      float nm = fmaxf(m, om);
      s = s * __expf(m - nm) + os * __expf(om - nm);
      m = nm;
    }
    if (mn == 0) {
      int row = rowBase + wave * 16 + v + 8 * kh;
      stats[2 * row + 0] = m;
      stats[2 * row + 1] = s;
    }
  }
}

// ---------------------------------------------------------------------------
// Kernel D: gather the 48 needed softmax columns per (b,s) row.
// ---------------------------------------------------------------------------
__global__ __launch_bounds__(64) void gather_emission(
    const _Float16* __restrict__ H16, const _Float16* __restrict__ WvT,
    const float* __restrict__ bv, const float* __restrict__ stats,
    const int* __restrict__ sources, float* __restrict__ EM) {
  __shared__ float hrow[HDIM];
  const int bs = blockIdx.x;                    // 0..3071 = b*96 + s
  const int b = bs / SDIM, s = bs % SDIM;
  const int r = (s < JLEN) ? (b * JLEN + s)
                           : (BATCH * JLEN + b * JLEN + (s - JLEN));
  for (int k = threadIdx.x; k < HDIM; k += 64)
    hrow[k] = (float)H16[(size_t)r * HDIM + k];
  __syncthreads();
  const int i = threadIdx.x;
  if (i < ILEN) {
    int src = sources[b * ILEN + i];
    const _Float16* w = WvT + (size_t)src * HDIM;
    float acc = 0.f;
    for (int k = 0; k < HDIM; ++k) acc += hrow[k] * (float)w[k];
    float mx = stats[2 * r], sm = stats[2 * r + 1];
    EM[(size_t)b * SDIM * ILEN + s * ILEN + i] = __expf(acc + bv[src] - mx) / sm;
  }
}

// ---------------------------------------------------------------------------
// Kernel E: scaled HMM forward-backward.  One block per batch element,
// 96 threads (3 wave32).  Transition matrix staged in LDS once (36 KB),
// alphas in LDS (18 KB); ~56 KB total, well under the 320 KB WGP budget.
// ---------------------------------------------------------------------------
__global__ __launch_bounds__(96) void fwd_bwd(
    const float* __restrict__ EM, const float* __restrict__ Atr,
    const float* __restrict__ pi, float* __restrict__ out) {
  __shared__ float Ash[SDIM * SDIM];    // 36 KB
  __shared__ float alpha[ILEN][SDIM];   // 18 KB
  __shared__ float cvec[ILEN];
  __shared__ float cur[SDIM];
  __shared__ float tmp[SDIM];
  __shared__ float wred[3];
  const int b = blockIdx.x, s = threadIdx.x;
  const int lane = s & 31, wv = s >> 5;
  const float* __restrict__ Ab = Atr + (size_t)b * SDIM * SDIM;
  const float* __restrict__ Eb = EM + (size_t)b * SDIM * ILEN;
  float* __restrict__ G = out + (size_t)b * SDIM * ILEN;

  for (int idx = s; idx < SDIM * SDIM; idx += SDIM) Ash[idx] = Ab[idx];

  // t = 0
  float a0 = pi[b * SDIM + s] * Eb[s * ILEN + 0];
  {
    float x = a0;
    x += __shfl_xor(x, 1, 32);  x += __shfl_xor(x, 2, 32);
    x += __shfl_xor(x, 4, 32);  x += __shfl_xor(x, 8, 32);
    x += __shfl_xor(x, 16, 32);
    if (lane == 0) wred[wv] = x;
  }
  __syncthreads();
  float c0 = wred[0] + wred[1] + wred[2];
  a0 /= c0;
  alpha[0][s] = a0;
  cur[s] = a0;
  if (s == 0) cvec[0] = c0;
  __syncthreads();

  // forward
  for (int t = 1; t < ILEN; ++t) {
    float an = 0.f;
    for (int u = 0; u < SDIM; ++u) an += cur[u] * Ash[u * SDIM + s];
    an *= Eb[s * ILEN + t];
    float x = an;
    x += __shfl_xor(x, 1, 32);  x += __shfl_xor(x, 2, 32);
    x += __shfl_xor(x, 4, 32);  x += __shfl_xor(x, 8, 32);
    x += __shfl_xor(x, 16, 32);
    if (lane == 0) wred[wv] = x;
    __syncthreads();
    float ct = wred[0] + wred[1] + wred[2];
    an /= ct;
    __syncthreads();            // everyone done reading cur
    cur[s] = an;
    alpha[t][s] = an;
    if (s == 0) cvec[t] = ct;
    __syncthreads();
  }

  // backward + gamma
  G[s * ILEN + (ILEN - 1)] = alpha[ILEN - 1][s];   // beta_{I-1} = 1
  cur[s] = 1.f;
  __syncthreads();
  for (int t = ILEN - 2; t >= 0; --t) {
    tmp[s] = Eb[s * ILEN + (t + 1)] * cur[s];
    __syncthreads();
    float bn = 0.f;
    for (int u = 0; u < SDIM; ++u) bn += Ash[s * SDIM + u] * tmp[u];
    bn /= cvec[t + 1];
    G[s * ILEN + t] = alpha[t][s] * bn;
    __syncthreads();
    cur[s] = bn;
    __syncthreads();
  }

  if (s == 0) {
    float ll = 0.f;
    for (int t = 0; t < ILEN; ++t) ll += __logf(cvec[t]);
    out[(size_t)BATCH * SDIM * ILEN + b] = -ll;
  }
}

// ---------------------------------------------------------------------------
// Launch
// ---------------------------------------------------------------------------
extern "C" void kernel_launch(void* const* d_in, const int* in_sizes, int n_in,
                              void* d_out, int out_size, void* d_ws,
                              size_t ws_size, hipStream_t stream) {
  const float* emb  = (const float*)d_in[0];   // [VT, E]
  const float* Wb   = (const float*)d_in[1];   // [E*CTX, E]
  const float* Wh   = (const float*)d_in[2];   // [E, H]
  const float* bh   = (const float*)d_in[3];   // [H]
  const float* Wv   = (const float*)d_in[4];   // [H, VS]
  const float* bv   = (const float*)d_in[5];   // [VS]
  const float* Atr  = (const float*)d_in[6];   // [B, S, S]
  const float* pi   = (const float*)d_in[7];   // [B, S]
  const int*   ctxW = (const int*)d_in[8];     // [B, J*CTX]
  const int*   ctxN = (const int*)d_in[9];     // [B, J*CTX]
  const int*   src  = (const int*)d_in[10];    // [B, I]
  float* out = (float*)d_out;

  // workspace layout (all offsets 64B aligned)
  char* ws = (char*)d_ws;
  const size_t OFF_WVT  = 0;                                     // 16384000 B
  const size_t OFF_H16  = OFF_WVT + (size_t)VS * HDIM * 2;       // +1572864
  const size_t OFF_ST   = OFF_H16 + (size_t)NROWS * HDIM * 2;    // +24576
  const size_t OFF_EM   = OFF_ST + (size_t)NROWS * 2 * 4;        // +589824
  _Float16* WvT = (_Float16*)(ws + OFF_WVT);
  _Float16* H16 = (_Float16*)(ws + OFF_H16);
  float*    ST  = (float*)(ws + OFF_ST);
  float*    EM  = (float*)(ws + OFF_EM);

  convert_wv      <<<VS / 256, 256, 0, stream>>>(Wv, WvT);
  hidden_kernel   <<<NROWS, 128, 0, stream>>>(emb, Wb, Wh, bh, ctxW, ctxN, H16);
  vocab_gemm_stats<<<NROWS / 64, 128, 0, stream>>>(H16, WvT, bv, ST);
  gather_emission <<<BATCH * SDIM, 64, 0, stream>>>(H16, WvT, bv, ST, src, EM);
  fwd_bwd         <<<BATCH, 96, 0, stream>>>(EM, Atr, pi, out);
}
